// PtrNet2_83150566851378
// MI455X (gfx1250) — matrix operand order repaired and checked by reading, not compiled, not run on hardware
//
#include <hip/hip_runtime.h>
#include <hip/hip_bf16.h>

#define DEV __device__ __forceinline__

// ---------------------------------------------------------------------------
// Problem constants (from the reference): B=128, N=402, E=4, F0=16, G=64, H=256
#define BB   128
#define NN   402
#define EE   4
#define F0_  16
#define GG   64
#define HH   256
#define NM2  400                 // N-2 (ref rows)
#define MFLAT (BB * NN)          // 51456 = 16*3216, exact multiple of 16
#define LD0  288                 // cat0 row: 256 (g0) + 16 (x) + 16 pad
#define LD1  544                 // cat1 row: 256 (g1) + 256 (h0) + 16 (x) + 16 pad
#define LDSPITCH 72              // padded LDS row pitch (bf16 elems) to spread banks

typedef __attribute__((ext_vector_type(16))) __bf16 v16bf;
typedef __attribute__((ext_vector_type(8)))  float  v8f;

union Frag16 {
  v16bf  bf;
  __bf16 h[16];
  uint4  q[2];
};

// ---------------------------------------------------------------------------
// Generic batched GEMM:  C[b,e] = post( A[b,e] (MxK) @ B[b,e] (KxN) )
//   post: optional per-row scale (inv-degree), per-col bias, ReLU.
//   AMODE 0: A bf16, row-major, K physically padded (two b128 loads per k-step).
//   AMODE 1: A f32 (edges): unguarded vectorized main loop, clamped-select tail
//            for the final partial k-step only (scalar-uniform branch).
//   B: bf16 row-major, K and N physically padded; staged through LDS per k-step
//      (shared by all 4 waves of the block).
//   One wave -> 16x64 output tile via 4x v_wmma_f32_16x16x32_bf16 per k-step.
struct GemmP {
  const __bf16* Abf;
  const float*  Af32;
  long ldA, sAb, sAe;
  const __bf16* Bm;
  long ldB, sBb, sBe;
  __bf16* Cbf;
  float*  Cf32;
  long ldC, sCb, sCe;
  const float* bias;
  const float* invdeg;
  long sIb, sIe;
  int M, N, K, KvalidA, E, relu;
};

template<int AMODE>
__global__ __launch_bounds__(128) void gemm_wmma(GemmP P)
{
  __shared__ alignas(16) __bf16 ldsB[32 * LDSPITCH];

  const int lane  = threadIdx.x & 31;
  const int wave  = threadIdx.x >> 5;
  const int z     = blockIdx.z;
  const int b     = z / P.E;
  const int e     = z - b * P.E;
  const long m0   = ((long)blockIdx.y * 4 + wave) * 16;
  const long n0   = (long)blockIdx.x * 64;
  const int khalf = (lane >= 16) ? 8 : 0;   // A K-offset / C M-offset for high half-wave

  long arow = m0 + (lane & 15);
  if (arow >= P.M) arow = P.M - 1;          // clamp (those rows are masked on store)

  // cooperative B-stage mapping: thread t -> row t/4, 16-col segment t%4
  const int srow = threadIdx.x >> 2;
  const int sseg = (threadIdx.x & 3) << 4;
  uint4* ldst = (uint4*)&ldsB[srow * LDSPITCH + sseg];
  const __bf16* bsrc = P.Bm + (long)b * P.sBb + (long)e * P.sBe + n0
                            + (long)srow * P.ldB + sseg;
  const long bstep = 32 * P.ldB;
  const __bf16* lrow = &ldsB[khalf * LDSPITCH + (lane & 15)];  // lane reads B column N=lane&15, K rows khalf+0..7, 16+khalf+0..7

  const __bf16* Aptr = nullptr;
  const float*  Fptr = nullptr;
  if constexpr (AMODE == 0)
    Aptr = P.Abf + (long)b * P.sAb + (long)e * P.sAe + arow * P.ldA + khalf;
  else
    Fptr = P.Af32 + (long)b * P.sAb + (long)e * P.sAe + arow * P.ldA + khalf;

  const int kmain = (AMODE == 0) ? P.K : (P.KvalidA & ~31);

  v8f acc[4] = {};

  for (int kb = 0; kb < P.K; kb += 32) {
    __syncthreads();                                    // protect previous reads
    {   // stage B[kb..kb+31][n0..n0+63] -> LDS
      const uint4* s = (const uint4*)bsrc;
      ldst[0] = s[0];
      ldst[1] = s[1];
    }
    bsrc += bstep;

    // ---- A fragment: lane holds row m0+(lane&15); two contiguous 8-elem runs
    Frag16 a;
    if constexpr (AMODE == 0) {
      a.q[0] = *(const uint4*)(Aptr);
      a.q[1] = *(const uint4*)(Aptr + 16);
      Aptr += 32;
    } else {
      if (kb < kmain) {                                 // uniform scalar branch
        __builtin_prefetch(Fptr + 32, 0, 1);            // global_prefetch_b8
        #pragma unroll
        for (int vv = 0; vv < 4; ++vv) {
          const float2 f0 = *(const float2*)(Fptr + 2 * vv);
          const float2 f1 = *(const float2*)(Fptr + 16 + 2 * vv);
          a.h[2 * vv]         = (__bf16)f0.x;
          a.h[2 * vv + 1]     = (__bf16)f0.y;
          a.h[8 + 2 * vv]     = (__bf16)f1.x;
          a.h[8 + 2 * vv + 1] = (__bf16)f1.y;
        }
      } else {                                          // final partial k-step
        #pragma unroll
        for (int i = 0; i < 16; ++i) {
          const int kabs = kb + khalf + i + (i & 8);    // elems 0-7, then +16
          const int kc   = (kabs < P.KvalidA) ? kabs : (P.KvalidA - 1);
          const float fv = Fptr[kc - kb - khalf];
          a.h[i] = (__bf16)((kabs < P.KvalidA) ? fv : 0.f);
        }
      }
      Fptr += 32;
    }

    __syncthreads();                                    // B tile visible

    #pragma unroll
    for (int nt = 0; nt < 4; ++nt) {
      Frag16 bfr;
      #pragma unroll
      for (int i = 0; i < 16; ++i)
        bfr.h[i] = lrow[(i + (i & 8)) * LDSPITCH + nt * 16];
      acc[nt] = __builtin_amdgcn_wmma_f32_16x16x32_bf16(
          false, a.bf, false, bfr.bf, (short)0, acc[nt], false, false);
    }
  }

  // ---- epilogue: C layout lane = N, VGPR r -> M = r (+8 for high half-wave)
  const float* idg = P.invdeg ? (P.invdeg + (long)b * P.sIb + (long)e * P.sIe) : nullptr;
  #pragma unroll
  for (int nt = 0; nt < 4; ++nt) {
    const long n = n0 + nt * 16 + (lane & 15);
    #pragma unroll
    for (int r = 0; r < 8; ++r) {
      const long m = m0 + r + khalf;
      if (m < P.M) {
        float val = acc[nt][r];
        if (idg)    val *= idg[m];
        if (P.bias) val += P.bias[n];
        if (P.relu) val  = fmaxf(val, 0.f);
        const long ci = (long)b * P.sCb + (long)e * P.sCe + m * P.ldC + n;
        if (P.Cf32) P.Cf32[ci] = val;
        else        P.Cbf[ci]  = (__bf16)val;
      }
    }
  }
}

// ---------------------------------------------------------------------------
// Weight prep: f32 (Kin x N) -> bf16 (Kpad x N), zero-filled pad rows. batch via grid.y
__global__ __launch_bounds__(256) void wprep_kernel(const float* __restrict__ src,
                                                    __bf16* __restrict__ dst,
                                                    int Kin, int Kpad, int N,
                                                    long sStride, long dStride)
{
  const int e = blockIdx.y;
  const long i = (long)blockIdx.x * 256 + threadIdx.x;
  if (i >= (long)Kpad * N) return;
  const long k = i / N;
  const long n = i - k * N;
  const float v = (k < Kin) ? src[(long)e * sStride + k * (long)N + n] : 0.f;
  dst[(long)e * dStride + i] = (__bf16)v;
}

// Copy x (f32) into the concat buffers as bf16
__global__ __launch_bounds__(256) void copyx_kernel(const float* __restrict__ x,
                                                    __bf16* __restrict__ cat0,
                                                    __bf16* __restrict__ cat1)
{
  const long i = (long)blockIdx.x * 256 + threadIdx.x;   // over MFLAT*16
  if (i >= (long)MFLAT * F0_) return;
  const long row = i >> 4;
  const int  f   = (int)(i & 15);
  const __bf16 v = (__bf16)x[i];
  cat0[row * LD0 + 256 + f] = v;
  cat1[row * LD1 + 512 + f] = v;
}

// inv_deg[b,e,m] = 1 / (sum_n edges[b,e,m,n] + 1e-10); one wave per row
__global__ __launch_bounds__(256) void deg_kernel(const float* __restrict__ edges,
                                                  float* __restrict__ invdeg, int rows)
{
  const int lane = threadIdx.x & 31;
  const int wrow = blockIdx.x * (blockDim.x >> 5) + (threadIdx.x >> 5);
  if (wrow >= rows) return;
  const float* p = edges + (long)wrow * NN;
  float s = 0.f;
  for (int n = lane; n < NN; n += 32) s += p[n];
  #pragma unroll
  for (int off = 16; off > 0; off >>= 1) s += __shfl_down(s, off, 32);
  if (lane == 0) invdeg[wrow] = 1.0f / (s + 1e-10f);
}

// query[b,h] = mean_n h1[b,n,h]
__global__ __launch_bounds__(256) void mean_kernel(const __bf16* __restrict__ h1,
                                                   float* __restrict__ query)
{
  const int b = blockIdx.x, h = threadIdx.x;
  const __bf16* p = h1 + (long)b * NN * HH + h;
  float s = 0.f;
  for (int n = 0; n < NN; ++n) s += (float)p[(long)n * HH];
  query[b * HH + h] = s * (1.0f / (float)NN);
}

// u1[b,h] = query[b] . Wq[:,h] + bq[h]
__global__ __launch_bounds__(256) void u1_kernel(const float* __restrict__ query,
                                                 const float* __restrict__ Wq,
                                                 const float* __restrict__ bq,
                                                 float* __restrict__ u1)
{
  const int b = blockIdx.x, h = threadIdx.x;
  const float* q = query + b * HH;
  float s = bq[h];
  for (int k = 0; k < HH; ++k) s += q[k] * Wq[k * HH + h];
  u1[b * HH + h] = s;
}

// u[b,n] = sum_h Vec[h] * tanh(u1[b,h] + u2[b,n,h])
__global__ __launch_bounds__(256) void u_kernel(const float* __restrict__ u1,
                                                const float* __restrict__ u2,
                                                const float* __restrict__ Vec,
                                                float* __restrict__ u)
{
  const int n = blockIdx.x, b = blockIdx.y, h = threadIdx.x;
  __shared__ float red[256];
  red[h] = Vec[h] * tanhf(u1[b * HH + h] + u2[((long)b * NM2 + n) * HH + h]);
  __syncthreads();
  #pragma unroll
  for (int s = 128; s > 0; s >>= 1) { if (h < s) red[h] += red[h + s]; __syncthreads(); }
  if (h == 0) u[b * NM2 + n] = red[0];
}

// softmax over n=0..399 (in place on u)
__global__ __launch_bounds__(256) void softmax_kernel(float* __restrict__ u)
{
  const int b = blockIdx.x, t = threadIdx.x;
  __shared__ float red[256];
  float m = -1e30f;
  for (int n = t; n < NM2; n += 256) m = fmaxf(m, u[b * NM2 + n]);
  red[t] = m; __syncthreads();
  #pragma unroll
  for (int s = 128; s > 0; s >>= 1) { if (t < s) red[t] = fmaxf(red[t], red[t + s]); __syncthreads(); }
  const float mx = red[0]; __syncthreads();
  float sum = 0.f;
  for (int n = t; n < NM2; n += 256) {
    const float ev = __expf(u[b * NM2 + n] - mx);
    u[b * NM2 + n] = ev; sum += ev;
  }
  red[t] = sum; __syncthreads();
  #pragma unroll
  for (int s = 128; s > 0; s >>= 1) { if (t < s) red[t] += red[t + s]; __syncthreads(); }
  const float inv = 1.0f / red[0];
  for (int n = t; n < NM2; n += 256) u[b * NM2 + n] *= inv;
}

// query[b,h] = sum_n a[b,n] * h1[b,n,h]
__global__ __launch_bounds__(256) void qup_kernel(const float* __restrict__ a,
                                                  const __bf16* __restrict__ h1,
                                                  float* __restrict__ query)
{
  const int b = blockIdx.x, h = threadIdx.x;
  const __bf16* p = h1 + (long)b * NN * HH + h;
  float s = 0.f;
  for (int n = 0; n < NM2; ++n) s += a[b * NM2 + n] * (float)p[(long)n * HH];
  query[b * HH + h] = s;
}

// pred[b] = sum_h relu(query[b].fc1[:,h]) * fc2[h]
__global__ __launch_bounds__(256) void pred_kernel(const float* __restrict__ query,
                                                   const float* __restrict__ fc1,
                                                   const float* __restrict__ fc2,
                                                   float* __restrict__ out)
{
  const int b = blockIdx.x, h = threadIdx.x;
  const float* q = query + b * HH;
  float s = 0.f;
  for (int k = 0; k < HH; ++k) s += q[k] * fc1[k * HH + h];
  s = fmaxf(s, 0.f) * fc2[h];
  __shared__ float red[256];
  red[h] = s; __syncthreads();
  #pragma unroll
  for (int st = 128; st > 0; st >>= 1) { if (h < st) red[h] += red[h + st]; __syncthreads(); }
  if (h == 0) out[b] = red[0];
}

// ---------------------------------------------------------------------------
static inline void launch_gemm(const GemmP& p, int gz, hipStream_t s)
{
  dim3 grid((unsigned)((p.N + 63) / 64), (unsigned)((p.M + 63) / 64), (unsigned)gz);
  if (p.Af32) gemm_wmma<1><<<grid, dim3(128), 0, s>>>(p);
  else        gemm_wmma<0><<<grid, dim3(128), 0, s>>>(p);
}

extern "C" void kernel_launch(void* const* d_in, const int* in_sizes, int n_in,
                              void* d_out, int out_size, void* d_ws, size_t ws_size,
                              hipStream_t stream)
{
  (void)in_sizes; (void)n_in; (void)out_size; (void)ws_size;
  const float* x       = (const float*)d_in[0];
  const float* edges   = (const float*)d_in[1];
  const float* gcrn1_W = (const float*)d_in[2];
  const float* ne0_W1  = (const float*)d_in[3];
  const float* ne0_b1  = (const float*)d_in[4];
  const float* ne0_W2  = (const float*)d_in[5];
  const float* ne0_b2  = (const float*)d_in[6];
  const float* gcrn2_W = (const float*)d_in[7];
  const float* ne1_W1  = (const float*)d_in[8];
  const float* ne1_b1  = (const float*)d_in[9];
  const float* ne1_W2  = (const float*)d_in[10];
  const float* ne1_b2  = (const float*)d_in[11];
  const float* Wq_W    = (const float*)d_in[12];
  const float* Wq_b    = (const float*)d_in[13];
  const float* Wref_W  = (const float*)d_in[14];
  const float* Wref_b  = (const float*)d_in[15];
  const float* Vec     = (const float*)d_in[16];
  const float* fc1     = (const float*)d_in[17];
  const float* fc2     = (const float*)d_in[18];

  // ---- workspace carve-out (256B aligned)
  char* w = (char*)d_ws;
  auto alloc = [&](size_t bytes) -> char* {
    char* p = w; w += (bytes + 255) & ~(size_t)255; return p;
  };
  float*   invdeg = (float*)  alloc((size_t)BB * EE * NN * 4);          // 0.8 MB
  __bf16*  wbuf   = (__bf16*) alloc((size_t)BB * EE * 416 * GG * 2);    // 27 MB (xw, then hw)
  __bf16*  cat0   = (__bf16*) alloc((size_t)MFLAT * LD0 * 2);           // 30 MB
  __bf16*  cat1   = (__bf16*) alloc((size_t)MFLAT * LD1 * 2);           // 56 MB
  __bf16*  bufT   = (__bf16*) alloc((size_t)MFLAT * HH * 2);            // 26 MB
  __bf16*  h1     = (__bf16*) alloc((size_t)MFLAT * HH * 2);            // 26 MB
  float*   u2     = (float*)  alloc((size_t)BB * NM2 * HH * 4);         // 52 MB
  __bf16*  W1a    = (__bf16*) alloc((size_t)288 * 256 * 2);
  __bf16*  W2a    = (__bf16*) alloc((size_t)256 * 256 * 2);
  __bf16*  W1b    = (__bf16*) alloc((size_t)544 * 256 * 2);
  __bf16*  W2b    = (__bf16*) alloc((size_t)256 * 256 * 2);
  __bf16*  Wrefb  = (__bf16*) alloc((size_t)256 * 256 * 2);
  __bf16*  gW1    = (__bf16*) alloc((size_t)EE * 32 * 64 * 2);
  __bf16*  gW2    = (__bf16*) alloc((size_t)EE * 256 * 64 * 2);
  float*   query  = (float*)  alloc((size_t)BB * HH * 4);
  float*   u1v    = (float*)  alloc((size_t)BB * HH * 4);
  float*   ubuf   = (float*)  alloc((size_t)BB * NM2 * 4);

  // ---- re-zero padded regions every call (ws is poisoned by the harness)
  hipMemsetAsync(wbuf, 0, (size_t)BB * EE * 416 * GG * 2, stream);
  hipMemsetAsync(cat0, 0, (size_t)MFLAT * LD0 * 2, stream);
  hipMemsetAsync(cat1, 0, (size_t)MFLAT * LD1 * 2, stream);

  // ---- bf16 weight prep (K padded)
  auto wp = [&](const float* src, __bf16* dst, int Kin, int Kpad, int N,
                int batch, long ss, long ds) {
    dim3 g((unsigned)(((long)Kpad * N + 255) / 256), (unsigned)batch);
    wprep_kernel<<<g, dim3(256), 0, stream>>>(src, dst, Kin, Kpad, N, ss, ds);
  };
  wp(ne0_W1,  W1a,  272, 288, 256, 1, 0, 0);
  wp(ne0_W2,  W2a,  256, 256, 256, 1, 0, 0);
  wp(ne1_W1,  W1b,  528, 544, 256, 1, 0, 0);
  wp(ne1_W2,  W2b,  256, 256, 256, 1, 0, 0);
  wp(Wref_W,  Wrefb,256, 256, 256, 1, 0, 0);
  wp(gcrn1_W, gW1,   16,  32,  64, EE, 16L * 64,  32L * 64);
  wp(gcrn2_W, gW2,  256, 256,  64, EE, 256L * 64, 256L * 64);

  copyx_kernel<<<dim3((MFLAT * F0_ + 255) / 256), dim3(256), 0, stream>>>(x, cat0, cat1);
  deg_kernel<<<dim3((BB * EE * NN + 7) / 8), dim3(256), 0, stream>>>(edges, invdeg, BB * EE * NN);

  // GEMM1: xw[b,e] = x[b] @ gcrn1_W[e]   (A from cat0 x-columns, K padded to 32)
  { GemmP p{}; p.Abf = cat0 + 256; p.ldA = LD0; p.sAb = (long)NN * LD0; p.sAe = 0;
    p.Bm = gW1; p.ldB = 64; p.sBb = 0; p.sBe = 32L * 64;
    p.Cbf = wbuf; p.ldC = 64; p.sCb = (long)EE * 416 * 64; p.sCe = 416L * 64;
    p.M = NN; p.N = 64; p.K = 32; p.KvalidA = 32; p.E = EE; p.relu = 0;
    launch_gemm(p, BB * EE, stream); }

  // GEMM2: g0 = relu(invdeg * edges @ xw) -> cat0[:, e*64 .. ]
  { GemmP p{}; p.Af32 = edges; p.ldA = NN; p.sAb = (long)EE * NN * NN; p.sAe = (long)NN * NN;
    p.Bm = wbuf; p.ldB = 64; p.sBb = (long)EE * 416 * 64; p.sBe = 416L * 64;
    p.Cbf = cat0; p.ldC = LD0; p.sCb = (long)NN * LD0; p.sCe = 64;
    p.invdeg = invdeg; p.sIb = (long)EE * NN; p.sIe = NN;
    p.M = NN; p.N = 64; p.K = 416; p.KvalidA = NN; p.E = EE; p.relu = 1;
    launch_gemm(p, BB * EE, stream); }

  // GEMM3: t = relu(cat0 @ ne0_W1 + b1)
  { GemmP p{}; p.Abf = cat0; p.ldA = LD0; p.Bm = W1a; p.ldB = 256;
    p.Cbf = bufT; p.ldC = 256; p.bias = ne0_b1;
    p.M = MFLAT; p.N = 256; p.K = 288; p.KvalidA = 288; p.E = 1; p.relu = 1;
    launch_gemm(p, 1, stream); }

  // GEMM4: h0 = relu(t @ ne0_W2 + b2) -> cat1[:, 256..511]
  { GemmP p{}; p.Abf = bufT; p.ldA = 256; p.Bm = W2a; p.ldB = 256;
    p.Cbf = cat1 + 256; p.ldC = LD1; p.bias = ne0_b2;
    p.M = MFLAT; p.N = 256; p.K = 256; p.KvalidA = 256; p.E = 1; p.relu = 1;
    launch_gemm(p, 1, stream); }

  // GEMM5: hw[b,e] = h0[b] @ gcrn2_W[e]
  { GemmP p{}; p.Abf = cat1 + 256; p.ldA = LD1; p.sAb = (long)NN * LD1; p.sAe = 0;
    p.Bm = gW2; p.ldB = 64; p.sBb = 0; p.sBe = 256L * 64;
    p.Cbf = wbuf; p.ldC = 64; p.sCb = (long)EE * 416 * 64; p.sCe = 416L * 64;
    p.M = NN; p.N = 64; p.K = 256; p.KvalidA = 256; p.E = EE; p.relu = 0;
    launch_gemm(p, BB * EE, stream); }

  // GEMM6: g1 = relu(invdeg * edges @ hw) -> cat1[:, e*64 .. ]
  { GemmP p{}; p.Af32 = edges; p.ldA = NN; p.sAb = (long)EE * NN * NN; p.sAe = (long)NN * NN;
    p.Bm = wbuf; p.ldB = 64; p.sBb = (long)EE * 416 * 64; p.sBe = 416L * 64;
    p.Cbf = cat1; p.ldC = LD1; p.sCb = (long)NN * LD1; p.sCe = 64;
    p.invdeg = invdeg; p.sIb = (long)EE * NN; p.sIe = NN;
    p.M = NN; p.N = 64; p.K = 416; p.KvalidA = NN; p.E = EE; p.relu = 1;
    launch_gemm(p, BB * EE, stream); }

  // GEMM7: t1 = relu(cat1 @ ne1_W1 + b1)
  { GemmP p{}; p.Abf = cat1; p.ldA = LD1; p.Bm = W1b; p.ldB = 256;
    p.Cbf = bufT; p.ldC = 256; p.bias = ne1_b1;
    p.M = MFLAT; p.N = 256; p.K = 544; p.KvalidA = 544; p.E = 1; p.relu = 1;
    launch_gemm(p, 1, stream); }

  // GEMM8: h1 = relu(t1 @ ne1_W2 + b2)
  { GemmP p{}; p.Abf = bufT; p.ldA = 256; p.Bm = W2b; p.ldB = 256;
    p.Cbf = h1; p.ldC = 256; p.bias = ne1_b2;
    p.M = MFLAT; p.N = 256; p.K = 256; p.KvalidA = 256; p.E = 1; p.relu = 1;
    launch_gemm(p, 1, stream); }

  // GEMM9: u2 = ref @ Wref + bref   (ref = h1[:, :-2], f32 output)
  { GemmP p{}; p.Abf = h1; p.ldA = 256; p.sAb = (long)NN * 256; p.sAe = 0;
    p.Bm = Wrefb; p.ldB = 256;
    p.Cf32 = u2; p.ldC = 256; p.sCb = (long)NM2 * 256; p.bias = Wref_b;
    p.M = NM2; p.N = 256; p.K = 256; p.KvalidA = 256; p.E = 1; p.relu = 0;
    launch_gemm(p, BB, stream); }

  // ---- attention decode (3 process iterations)
  mean_kernel<<<dim3(BB), dim3(HH), 0, stream>>>(h1, query);
  for (int it = 0; it < 3; ++it) {
    u1_kernel<<<dim3(BB), dim3(HH), 0, stream>>>(query, Wq_W, Wq_b, u1v);
    u_kernel<<<dim3(NM2, BB), dim3(HH), 0, stream>>>(u1v, u2, Vec, ubuf);
    softmax_kernel<<<dim3(BB), dim3(256), 0, stream>>>(ubuf);
    qup_kernel<<<dim3(BB), dim3(HH), 0, stream>>>(ubuf, h1, query);
  }
  pred_kernel<<<dim3(BB), dim3(HH), 0, stream>>>(query, fc1, fc2, (float*)d_out);
}
